// WaveNetX_44951127720119
// MI455X (gfx1250) — compile-verified, matrix operand order checked by reading, not run on hardware
//
#include <hip/hip_runtime.h>

#define Bb  64
#define Ll  2048
#define Ee  128
#define Cc  64
#define Ffc 128
#define Oo  30
#define NDl 9
#define GEe 1024
#define NT  128   // positions per block tile

typedef __attribute__((ext_vector_type(16))) _Float16 v16h;
typedef __attribute__((ext_vector_type(8)))  _Float16 v8h;
typedef __attribute__((ext_vector_type(8)))  float    v8f;

// Fragment loader for 16-bit WMMA A/B operands from an LDS tile stored
// row-major [row][k] with `stride` halfs per row (stride % 8 == 0, base 16B
// aligned). Per-lane layout (ISA 7.12.2, 16-bit A 16x32, B mirrored):
//   halves 0..7  -> k = k0 + (lane<16 ? 0 : 8) + 0..7
//   halves 8..15 -> k = k0 + 16 + (lane<16 ? 0 : 8) + 0..7
static __device__ __forceinline__ v16h load_frag(const _Float16* base, int row,
                                                 int stride, int k0, int lane) {
  const int kb = k0 + ((lane & 16) ? 8 : 0);
  const _Float16* p = base + row * stride + kb;
  v8h lo = *(const v8h*)(p);
  v8h hi = *(const v8h*)(p + 16);
  v16h r;
#pragma unroll
  for (int i = 0; i < 8; ++i) { r[i] = lo[i]; r[i + 8] = hi[i]; }
  return r;
}

static __device__ __forceinline__ v8f wmma16(v16h a, v16h b, v8f c) {
  return __builtin_amdgcn_wmma_f32_16x16x32_f16(false, a, false, b,
                                                (short)0, c, false, false);
}

// ---------------------------------------------------------------------------
// Weight conversion: pack all weights into f16 GEMM layouts in workspace.
//   WCh[k][c][e]          (2 x 64 x 128)  from w_causal[c][e][k]
//   WFh[i][k][co][ci]     (9 x 2 x 64 x 64) from w_f[i][co][ci][k]  (WGh same)
//   WRh[i][o][c]          (9 x 64 x 64)
//   W1h[f][c]             (128 x 64)
//   W2h[o][f]             (32 x 128, rows >= 30 zero)
// ---------------------------------------------------------------------------
__global__ void k_wconv(const float* __restrict__ wc, const float* __restrict__ wf,
                        const float* __restrict__ wg, const float* __restrict__ wr,
                        const float* __restrict__ w1, const float* __restrict__ w2,
                        _Float16* WCh, _Float16* WFh, _Float16* WGh,
                        _Float16* WRh, _Float16* W1h, _Float16* W2h) {
  int i = blockIdx.x * 256 + threadIdx.x;
  if (i < 16384) {                       // WCh
    int k = i >> 13, ce = i & 8191;
    WCh[i] = (_Float16)wc[ce * 2 + k];
  } else if (i < 16384 + 73728) {        // WFh + WGh
    int j = i - 16384;
    int lyr = j >> 13; int k = (j >> 12) & 1; int co = (j >> 6) & 63; int ci = j & 63;
    size_t s = ((size_t)(lyr * 64 + co) * 64 + ci) * 2 + k;
    WFh[j] = (_Float16)wf[s];
    WGh[j] = (_Float16)wg[s];
  } else if (i < 16384 + 73728 + 36864) { // WRh
    int j = i - (16384 + 73728);
    WRh[j] = (_Float16)wr[j];
  } else if (i < 16384 + 73728 + 36864 + 8192) { // W1h
    int j = i - (16384 + 73728 + 36864);
    W1h[j] = (_Float16)w1[j];
  } else if (i < 16384 + 73728 + 36864 + 8192 + 4096) { // W2h (padded)
    int j = i - (16384 + 73728 + 36864 + 8192);
    int o = j >> 7;
    W2h[j] = (o < Oo) ? (_Float16)w2[j] : (_Float16)0.0f;
  }
}

// emb_g[b][g*E+e] = table[gin[b][g]][e], with padding_idx==0 -> 0
__global__ void k_embg(const int* __restrict__ gin, const float* __restrict__ table,
                       float* __restrict__ embg) {
  int i = blockIdx.x * 256 + threadIdx.x;
  if (i >= Bb * GEe) return;
  int b = i >> 10; int j = i & 1023; int g = j >> 7; int e = j & 127;
  int t = gin[b * 8 + g];
  embg[i] = (t > 0) ? table[(size_t)t * Ee + e] : 0.0f;
}

// H[(2*i+s)][b][c] = emb_g[b] . wl_{f,g}[i][c] + bl  (tiny: 75 MFLOP)
__global__ void k_h(const float* __restrict__ embg,
                    const float* __restrict__ wlf, const float* __restrict__ wlg,
                    const float* __restrict__ blf, const float* __restrict__ blg,
                    float* __restrict__ H) {
  int idx = blockIdx.x * 256 + threadIdx.x;
  if (idx >= 18 * Bb * Cc) return;
  int c = idx & 63; int b = (idx >> 6) & 63; int i2 = idx >> 12;
  int lyr = i2 >> 1; int s = i2 & 1;
  const float* w = (s ? wlg : wlf) + ((size_t)lyr * Cc + c) * GEe;
  const float* x = embg + (size_t)b * GEe;
  float acc = (s ? blg : blf)[lyr * Cc + c];
  for (int k = 0; k < GEe; k += 4) {
    float4 wv = *(const float4*)(w + k);
    float4 xv = *(const float4*)(x + k);
    acc += wv.x * xv.x + wv.y * xv.y + wv.z * xv.z + wv.w * xv.w;
  }
  H[idx] = acc;
}

// ---------------------------------------------------------------------------
// Initial causal conv (E=128 -> C=64, K=2, d=1) fused with embedding lookup.
// Wave mapping: wave w -> (mtile = w&3, nhalf = w>>2); A fragments hoisted to
// registers (2 taps x 4 K-steps = 8 fragments, 64 VGPRs), reused across the
// wave's 4 N-subtiles. Writes res (ping 0, [B][L][C]) and zeroes skips.
// ---------------------------------------------------------------------------
union InitSmem {
  _Float16 Xe[NT + 1][Ee];   // rows n=0..128 <-> l = l0-1+n   (33 KB)
  float    SK[NT][Cc];       // f32 store staging               (32 KB)
};

__global__ __launch_bounds__(256) void k_init(
    const int* __restrict__ tok, const float* __restrict__ table,
    const _Float16* __restrict__ WCh, const float* __restrict__ bc,
    float* __restrict__ rout, float* __restrict__ skips) {
  __shared__ __align__(16) InitSmem u;
  __shared__ __align__(16) _Float16 sW[2][Cc][Ee];

  const int b = blockIdx.y, l0 = blockIdx.x * NT, tid = threadIdx.x;
  _Float16* sWp = &sW[0][0][0];
  for (int i = tid; i < 2 * Cc * Ee; i += 256) sWp[i] = WCh[i];

  for (int i4 = tid; i4 < (NT + 1) * (Ee / 4); i4 += 256) {
    int n = i4 >> 5; int e = (i4 & 31) * 4;
    int l = l0 - 1 + n;
    _Float16* p = &u.Xe[n][e];
    int t = (l >= 0) ? tok[b * Ll + l] : 0;
    if (t > 0) {
      float4 v = *(const float4*)(table + (size_t)t * Ee + e);
      p[0] = (_Float16)v.x; p[1] = (_Float16)v.y;
      p[2] = (_Float16)v.z; p[3] = (_Float16)v.w;
    } else {
      p[0] = p[1] = p[2] = p[3] = (_Float16)0.0f;
    }
  }
  __syncthreads();

  const int lane = tid & 31, wav = tid >> 5;
  const int m = wav & 3, nh = wav >> 2;
  const int col = lane & 15, rb = (lane & 16) ? 8 : 0;
  const int arow = m * 16 + col;

  // Hoist weight (A) fragments into registers: 2 taps x 4 K-steps.
  v16h aW0[4], aW1[4];
#pragma unroll
  for (int ks = 0; ks < 4; ++ks) {
    aW0[ks] = load_frag(sWp,           arow, Ee, ks * 32, lane);
    aW1[ks] = load_frag(sWp + Cc * Ee, arow, Ee, ks * 32, lane);
  }
  float initC[8];
#pragma unroll
  for (int r = 0; r < 8; ++r) initC[r] = bc[m * 16 + rb + r];

  v8f acc[4];
#pragma unroll
  for (int nt = 0; nt < 4; ++nt)
#pragma unroll
    for (int r = 0; r < 8; ++r) acc[nt][r] = initC[r];

#pragma unroll
  for (int nt = 0; nt < 4; ++nt) {
    const int nrow = nh * 64 + nt * 16 + col;
#pragma unroll
    for (int ks = 0; ks < 4; ++ks) {
      v16h Bd = load_frag(&u.Xe[0][0], nrow,     Ee, ks * 32, lane); // x[l-1]
      v16h Bc = load_frag(&u.Xe[0][0], nrow + 1, Ee, ks * 32, lane); // x[l]
      acc[nt] = wmma16(aW0[ks], Bd, acc[nt]);
      acc[nt] = wmma16(aW1[ks], Bc, acc[nt]);
    }
  }
  __syncthreads();  // all Xe reads done; safe to alias SK

#pragma unroll
  for (int nt = 0; nt < 4; ++nt)
#pragma unroll
    for (int r = 0; r < 8; ++r)
      u.SK[nh * 64 + nt * 16 + col][m * 16 + rb + r] = acc[nt][r];
  __syncthreads();

  float* dr = rout  + ((size_t)b * Ll + l0) * Cc;
  float* ds = skips + ((size_t)b * Ll + l0) * Cc;
  for (int i4 = tid; i4 < NT * Cc / 4; i4 += 256) {
    float4 v = *((const float4*)&u.SK[0][0] + i4);
    *((float4*)dr + i4) = v;
    float4 z = {0.0f, 0.0f, 0.0f, 0.0f};
    *((float4*)ds + i4) = z;
  }
}

// ---------------------------------------------------------------------------
// One dilated residual layer:
//   xf = Wf0 @ res[l-d] + Wf1 @ res[l] + bf + hf[b]
//   xg = Wg0 @ res[l-d] + Wg1 @ res[l] + bg + hg[b]
//   z  = tanh(xf) * sigmoid(xg)
//   skip = Wr @ z + br ; res_out = res_in + skip ; skips += skip
// Wave mapping: wave w -> (mtile = w&3, nhalf = w>>2). All 10 weight (A)
// fragments (F0/F1/G0/G1/R x 2 K-steps, 80 VGPRs) hoisted once per block and
// reused across 4 N-subtiles, so the steady-state LDS traffic is B-side only.
// ---------------------------------------------------------------------------
union LayerSmem {
  struct { _Float16 Xc[NT][Cc]; _Float16 Xd[NT][Cc]; } x;  // 32 KB
  float SK[NT][Cc];                                         // 32 KB
};

__global__ __launch_bounds__(256) void k_layer(
    const float* __restrict__ rin, float* __restrict__ rout,
    float* __restrict__ skips,
    const _Float16* __restrict__ WF, const _Float16* __restrict__ WG,
    const _Float16* __restrict__ WR,
    const float* __restrict__ bf, const float* __restrict__ bg,
    const float* __restrict__ br,
    const float* __restrict__ Hf, const float* __restrict__ Hg, int d) {
  __shared__ __align__(16) LayerSmem u;
  __shared__ __align__(16) _Float16 sW[5][Cc][Cc]; // F0,F1,G0,G1,R  (40 KB)
  __shared__ __align__(16) _Float16 Zs[NT][Cc];    // gated output   (16 KB)

  const int b = blockIdx.y, l0 = blockIdx.x * NT, tid = threadIdx.x;
  _Float16* sWp = &sW[0][0][0];

  for (int i = tid; i < Cc * Cc; i += 256) {
    sWp[0 * 4096 + i] = WF[i];
    sWp[1 * 4096 + i] = WF[4096 + i];
    sWp[2 * 4096 + i] = WG[i];
    sWp[3 * 4096 + i] = WG[4096 + i];
    sWp[4 * 4096 + i] = WR[i];
  }

  const float* src = rin + (size_t)b * Ll * Cc;
  for (int i4 = tid; i4 < NT * Cc / 4; i4 += 256) {
    int flat = i4 * 4;
    int n = flat >> 6;
    float4 vc = *(const float4*)(src + (size_t)(l0 + n) * Cc + (flat & 63));
    _Float16* pc = &u.x.Xc[0][0] + flat;
    pc[0] = (_Float16)vc.x; pc[1] = (_Float16)vc.y;
    pc[2] = (_Float16)vc.z; pc[3] = (_Float16)vc.w;
    int ld = l0 + n - d;
    _Float16* pd = &u.x.Xd[0][0] + flat;
    if (ld >= 0) {
      float4 vd = *(const float4*)(src + (size_t)ld * Cc + (flat & 63));
      pd[0] = (_Float16)vd.x; pd[1] = (_Float16)vd.y;
      pd[2] = (_Float16)vd.z; pd[3] = (_Float16)vd.w;
    } else {
      pd[0] = pd[1] = pd[2] = pd[3] = (_Float16)0.0f;
    }
  }
  __syncthreads();

  const int lane = tid & 31, wav = tid >> 5;
  const int m = wav & 3, nh = wav >> 2;
  const int col = lane & 15, rb = (lane & 16) ? 8 : 0;
  const int arow = m * 16 + col;

  // Hoisted A fragments: 5 weight matrices x 2 K-steps.
  v16h aWF0[2], aWF1[2], aWG0[2], aWG1[2], aWR[2];
#pragma unroll
  for (int ks = 0; ks < 2; ++ks) {
    const int k0 = ks * 32;
    aWF0[ks] = load_frag(sWp + 0 * 4096, arow, Cc, k0, lane);
    aWF1[ks] = load_frag(sWp + 1 * 4096, arow, Cc, k0, lane);
    aWG0[ks] = load_frag(sWp + 2 * 4096, arow, Cc, k0, lane);
    aWG1[ks] = load_frag(sWp + 3 * 4096, arow, Cc, k0, lane);
    aWR [ks] = load_frag(sWp + 4 * 4096, arow, Cc, k0, lane);
  }

  float initF[8], initG[8];
#pragma unroll
  for (int r = 0; r < 8; ++r) {
    int c = m * 16 + rb + r;
    initF[r] = bf[c] + Hf[b * Cc + c];
    initG[r] = bg[c] + Hg[b * Cc + c];
  }

  v8f aF[4], aG[4];
#pragma unroll
  for (int nt = 0; nt < 4; ++nt)
#pragma unroll
    for (int r = 0; r < 8; ++r) { aF[nt][r] = initF[r]; aG[nt][r] = initG[r]; }

#pragma unroll
  for (int nt = 0; nt < 4; ++nt) {
    const int nrow = nh * 64 + nt * 16 + col;
#pragma unroll
    for (int ks = 0; ks < 2; ++ks) {
      const int k0 = ks * 32;
      v16h Bc = load_frag(&u.x.Xc[0][0], nrow, Cc, k0, lane);
      v16h Bd = load_frag(&u.x.Xd[0][0], nrow, Cc, k0, lane);
      aF[nt] = wmma16(aWF0[ks], Bd, aF[nt]);
      aF[nt] = wmma16(aWF1[ks], Bc, aF[nt]);
      aG[nt] = wmma16(aWG0[ks], Bd, aG[nt]);
      aG[nt] = wmma16(aWG1[ks], Bc, aG[nt]);
    }
  }

  // gated activation -> Zs (f16, [n][c])
#pragma unroll
  for (int nt = 0; nt < 4; ++nt)
#pragma unroll
    for (int r = 0; r < 8; ++r) {
      float f = aF[nt][r], g = aG[nt][r];
      float z = tanhf(f) * (1.0f / (1.0f + __expf(-g)));
      Zs[nh * 64 + nt * 16 + col][m * 16 + rb + r] = (_Float16)z;
    }
  __syncthreads();  // Zs complete across mtiles; u.x reads done -> SK alias ok

  v8f aS[4];
  float initS[8];
#pragma unroll
  for (int r = 0; r < 8; ++r) initS[r] = br[m * 16 + rb + r];
#pragma unroll
  for (int nt = 0; nt < 4; ++nt)
#pragma unroll
    for (int r = 0; r < 8; ++r) aS[nt][r] = initS[r];

#pragma unroll
  for (int nt = 0; nt < 4; ++nt) {
    const int nrow = nh * 64 + nt * 16 + col;
#pragma unroll
    for (int ks = 0; ks < 2; ++ks) {
      v16h Bz = load_frag(&Zs[0][0], nrow, Cc, ks * 32, lane);
      aS[nt] = wmma16(aWR[ks], Bz, aS[nt]);
    }
  }

#pragma unroll
  for (int nt = 0; nt < 4; ++nt)
#pragma unroll
    for (int r = 0; r < 8; ++r)
      u.SK[nh * 64 + nt * 16 + col][m * 16 + rb + r] = aS[nt][r];
  __syncthreads();

  float* dr = rout  + ((size_t)b * Ll + l0) * Cc;
  float* dk = skips + ((size_t)b * Ll + l0) * Cc;
  const float* sr = src + (size_t)l0 * Cc;
  for (int i4 = tid; i4 < NT * Cc / 4; i4 += 256) {
    float4 sk = *((const float4*)&u.SK[0][0] + i4);
    float4 rv = *((const float4*)sr + i4);
    float4 sv = *((const float4*)dk + i4);
    rv.x += sk.x; rv.y += sk.y; rv.z += sk.z; rv.w += sk.w;
    sv.x += sk.x; sv.y += sk.y; sv.z += sk.z; sv.w += sk.w;
    *((float4*)dr + i4) = rv;
    *((float4*)dk + i4) = sv;
  }
}

// ---------------------------------------------------------------------------
// Head: relu(skips) -> fin1 (128x64) -> relu -> fin2 (32x128, 30 valid) -> out
// Output layout [B][O][L] f32.
// ---------------------------------------------------------------------------
union FinalSmem {
  _Float16 Xs[NT][Cc];   // 16 KB, relu(skips) [n][c]
  float    Os[32][NT];   // 16 KB, output staging [o][n]
};

__global__ __launch_bounds__(256) void k_final(
    const float* __restrict__ skips,
    const _Float16* __restrict__ W1, const _Float16* __restrict__ W2,
    const float* __restrict__ b1, const float* __restrict__ b2,
    float* __restrict__ out) {
  __shared__ __align__(16) FinalSmem u;
  __shared__ __align__(16) _Float16 sW1[Ffc][Cc];   // 16 KB
  __shared__ __align__(16) _Float16 sW2[32][Ffc];   //  8 KB
  __shared__ __align__(16) _Float16 Ys[NT][Ffc];    // 32 KB

  const int b = blockIdx.y, l0 = blockIdx.x * NT, tid = threadIdx.x;

  const float* src = skips + ((size_t)b * Ll + l0) * Cc;
  for (int i4 = tid; i4 < NT * Cc / 4; i4 += 256) {
    float4 v = *((const float4*)src + i4);
    _Float16* p = &u.Xs[0][0] + i4 * 4;
    p[0] = (_Float16)fmaxf(v.x, 0.0f); p[1] = (_Float16)fmaxf(v.y, 0.0f);
    p[2] = (_Float16)fmaxf(v.z, 0.0f); p[3] = (_Float16)fmaxf(v.w, 0.0f);
  }
  for (int i = tid; i < Ffc * Cc; i += 256) (&sW1[0][0])[i] = W1[i];
  for (int i = tid; i < 32 * Ffc; i += 256) (&sW2[0][0])[i] = W2[i];
  __syncthreads();

  const int lane = tid & 31, wav = tid >> 5;
  const int n0 = wav * 16, col = lane & 15, rb = (lane & 16) ? 8 : 0;

  // fin1: M=128, K=64
  v8f a1[8];
#pragma unroll
  for (int m = 0; m < 8; ++m)
#pragma unroll
    for (int r = 0; r < 8; ++r) a1[m][r] = b1[m * 16 + rb + r];
#pragma unroll
  for (int k0 = 0; k0 < Cc; k0 += 32) {
    v16h Bx = load_frag(&u.Xs[0][0], n0 + col, Cc, k0, lane);
#pragma unroll
    for (int m = 0; m < 8; ++m) {
      v16h A = load_frag(&sW1[0][0], m * 16 + col, Cc, k0, lane);
      a1[m] = wmma16(A, Bx, a1[m]);
    }
  }
#pragma unroll
  for (int m = 0; m < 8; ++m)
#pragma unroll
    for (int r = 0; r < 8; ++r)
      Ys[n0 + col][m * 16 + rb + r] = (_Float16)fmaxf(a1[m][r], 0.0f);
  __syncthreads();  // all Xs reads done -> safe to alias u.Os

  // fin2: M=32 (30 valid), K=128
  v8f a2[2];
#pragma unroll
  for (int m = 0; m < 2; ++m)
#pragma unroll
    for (int r = 0; r < 8; ++r) {
      int o = m * 16 + rb + r;
      a2[m][r] = (o < Oo) ? b2[o] : 0.0f;
    }
#pragma unroll
  for (int k0 = 0; k0 < Ffc; k0 += 32) {
    v16h By = load_frag(&Ys[0][0], n0 + col, Ffc, k0, lane);
#pragma unroll
    for (int m = 0; m < 2; ++m) {
      v16h A = load_frag(&sW2[0][0], m * 16 + col, Ffc, k0, lane);
      a2[m] = wmma16(A, By, a2[m]);
    }
  }
#pragma unroll
  for (int m = 0; m < 2; ++m)
#pragma unroll
    for (int r = 0; r < 8; ++r)
      u.Os[m * 16 + rb + r][n0 + col] = a2[m][r];
  __syncthreads();

  for (int i = tid; i < Oo * NT; i += 256) {
    int o = i >> 7, n = i & 127;
    out[((size_t)b * Oo + o) * Ll + l0 + n] = u.Os[o][n];
  }
}

// ---------------------------------------------------------------------------
extern "C" void kernel_launch(void* const* d_in, const int* in_sizes, int n_in,
                              void* d_out, int out_size, void* d_ws, size_t ws_size,
                              hipStream_t stream) {
  (void)in_sizes; (void)n_in; (void)out_size; (void)ws_size;
  const int*   tok      = (const int*)  d_in[0];
  const int*   gin      = (const int*)  d_in[1];
  const float* table    = (const float*)d_in[2];
  const float* w_causal = (const float*)d_in[3];
  const float* b_causal = (const float*)d_in[4];
  const float* w_f      = (const float*)d_in[5];
  const float* b_f      = (const float*)d_in[6];
  const float* w_g      = (const float*)d_in[7];
  const float* b_g      = (const float*)d_in[8];
  const float* wl_f     = (const float*)d_in[9];
  const float* bl_f     = (const float*)d_in[10];
  const float* wl_g     = (const float*)d_in[11];
  const float* bl_g     = (const float*)d_in[12];
  const float* w_res    = (const float*)d_in[13];
  const float* b_res    = (const float*)d_in[14];
  const float* w_fin1   = (const float*)d_in[15];
  const float* b_fin1   = (const float*)d_in[16];
  const float* w_fin2   = (const float*)d_in[17];
  const float* b_fin2   = (const float*)d_in[18];
  float* out = (float*)d_out;

  char* ws = (char*)d_ws;
  size_t off = 0;
  auto carve = [&](size_t bytes) -> char* {
    char* p = ws + off;
    off += (bytes + 255) & ~(size_t)255;
    return p;
  };
  float* resf0 = (float*)carve((size_t)Bb * Ll * Cc * 4);
  float* resf1 = (float*)carve((size_t)Bb * Ll * Cc * 4);
  float* skips = (float*)carve((size_t)Bb * Ll * Cc * 4);
  float* embg  = (float*)carve((size_t)Bb * GEe * 4);
  float* H     = (float*)carve((size_t)18 * Bb * Cc * 4);
  _Float16* WCh = (_Float16*)carve((size_t)2 * Cc * Ee * 2);
  _Float16* WFh = (_Float16*)carve((size_t)NDl * 2 * Cc * Cc * 2);
  _Float16* WGh = (_Float16*)carve((size_t)NDl * 2 * Cc * Cc * 2);
  _Float16* WRh = (_Float16*)carve((size_t)NDl * Cc * Cc * 2);
  _Float16* W1h = (_Float16*)carve((size_t)Ffc * Cc * 2);
  _Float16* W2h = (_Float16*)carve((size_t)32 * Ffc * 2);

  k_wconv<<<544, 256, 0, stream>>>(w_causal, w_f, w_g, w_res, w_fin1, w_fin2,
                                   WCh, WFh, WGh, WRh, W1h, W2h);
  k_embg<<<(Bb * GEe + 255) / 256, 256, 0, stream>>>(gin, table, embg);
  k_h<<<(18 * Bb * Cc + 255) / 256, 256, 0, stream>>>(embg, wl_f, wl_g, bl_f, bl_g, H);

  dim3 grid(Ll / NT, Bb);
  k_init<<<grid, 256, 0, stream>>>(tok, table, WCh, b_causal, resf0, skips);

  float* rin = resf0;
  float* rout = resf1;
  for (int i = 0; i < NDl; ++i) {
    int d = 2 << i;  // 2^(i+1)
    k_layer<<<grid, 256, 0, stream>>>(
        rin, rout, skips,
        WFh + (size_t)i * 2 * Cc * Cc, WGh + (size_t)i * 2 * Cc * Cc,
        WRh + (size_t)i * Cc * Cc,
        b_f + i * Cc, b_g + i * Cc, b_res + i * Cc,
        H + (size_t)(2 * i) * Bb * Cc, H + (size_t)(2 * i + 1) * Bb * Cc, d);
    float* t = rin; rin = rout; rout = t;
  }
  k_final<<<grid, 256, 0, stream>>>(skips, W1h, W2h, b_fin1, b_fin2, out);
}
